// LGA_26963804685123
// MI455X (gfx1250) — compile-verified, matrix operand order checked by reading, not run on hardware
//
#include <hip/hip_runtime.h>
#include <hip/hip_bf16.h>

// ---------------------------------------------------------------------------
// Problem constants (match reference)
// ---------------------------------------------------------------------------
#define BB      8
#define NN      2048
#define DD      192
#define GG      512
#define SS      32
#define OUTC    384
#define HC      192           // OUT/2
#define TWO_D   384           // 2*D
#define NGROUP  (BB*GG)       // 4096
#define CENTERS_OUT (BB*GG*3) // 12288 floats at start of d_out

typedef __bf16 bf16_t;
typedef __attribute__((ext_vector_type(16))) __bf16 v16bf;
typedef __attribute__((ext_vector_type(8)))  __bf16 v8bf;
typedef __attribute__((ext_vector_type(8)))  float  v8f;

// ---------------------------------------------------------------------------
// WMMA helpers (CDNA5 wave32, 16x16x32 bf16 -> f32)
// A tile 16x32 and B tile 32x16 share the same per-lane layout (ISA 7.12.2):
//   lane = {m|n} in [0,16) for both halves, half = lane>>4
//   elems 0..7  : K = k0 + 8*half     .. +7   (16B contiguous)
//   elems 8..15 : K = k0 + 16 + 8*half .. +7  (16B contiguous)
// ---------------------------------------------------------------------------
__device__ __forceinline__ v16bf load_frag(const bf16_t* __restrict__ base,
                                           int ld, int r0, int k0, int lane) {
  const int m = lane & 15, h = lane >> 4;
  const bf16_t* p = base + (size_t)(r0 + m) * ld + k0 + 8 * h;
  v8bf lo = *(const v8bf*)(p);
  v8bf hi = *(const v8bf*)(p + 16);
  v16bf f;
#pragma unroll
  for (int i = 0; i < 8; ++i) { f[i] = lo[i]; f[8 + i] = hi[i]; }
  return f;
}

__device__ __forceinline__ v8f wmma_bf16(v16bf a, v16bf b, v8f c) {
  return __builtin_amdgcn_wmma_f32_16x16x32_bf16(false, a, false, b,
                                                 (short)0, c, false, false);
}

__device__ __forceinline__ v8f vzero8() {
  v8f z = {0.f, 0.f, 0.f, 0.f, 0.f, 0.f, 0.f, 0.f};
  return z;
}

// ---------------------------------------------------------------------------
// K0: weights fp32 -> bf16
// ---------------------------------------------------------------------------
__global__ __launch_bounds__(256) void k_prep(
    const float* __restrict__ W1, const float* __restrict__ Wa,
    const float* __restrict__ Wb, bf16_t* __restrict__ W1b,
    bf16_t* __restrict__ Wab, bf16_t* __restrict__ Wbb) {
  int i = blockIdx.x * 256 + threadIdx.x;
  if (i < OUTC * TWO_D) W1b[i] = (bf16_t)W1[i];
  if (i < HC * OUTC) { Wab[i] = (bf16_t)Wa[i]; Wbb[i] = (bf16_t)Wb[i]; }
}

// ---------------------------------------------------------------------------
// K1: farthest point sampling, one block per batch
// ---------------------------------------------------------------------------
__global__ __launch_bounds__(256) void k_fps(
    const float* __restrict__ pts, float* __restrict__ centers,
    float* __restrict__ out_centers) {
  __shared__ float sp[NN * 3];
  __shared__ float sd[NN];
  __shared__ float rv[256];
  __shared__ int   ri[256];

  const int t = threadIdx.x;
  const int b = blockIdx.x;
  const float* P = pts + (size_t)b * NN * 3;
  const float INF = __builtin_inff();

  for (int i = t; i < NN * 3; i += 256) sp[i] = P[i];
  for (int n = t; n < NN; n += 256) sd[n] = INF;
  __syncthreads();

  int last = 0;
  for (int g = 0; g < GG; ++g) {
    float cx = sp[last * 3 + 0], cy = sp[last * 3 + 1], cz = sp[last * 3 + 2];
    if (t == 0) {
      size_t o = ((size_t)b * GG + g) * 3;
      centers[o + 0] = cx; centers[o + 1] = cy; centers[o + 2] = cz;
      out_centers[o + 0] = cx; out_centers[o + 1] = cy; out_centers[o + 2] = cz;
    }
    float bv = -INF; int bi = 0x7fffffff;
    for (int n = t; n < NN; n += 256) {
      float dx = sp[n * 3 + 0] - cx;
      float dy = sp[n * 3 + 1] - cy;
      float dz = sp[n * 3 + 2] - cz;
      float d = dx * dx + dy * dy + dz * dz;
      float nd = fminf(sd[n], d);
      sd[n] = nd;
      if (nd > bv || (nd == bv && n < bi)) { bv = nd; bi = n; }
    }
    rv[t] = bv; ri[t] = bi;
    __syncthreads();
    for (int off = 128; off > 0; off >>= 1) {
      if (t < off) {
        float ov = rv[t + off]; int oi = ri[t + off];
        if (ov > rv[t] || (ov == rv[t] && oi < ri[t])) { rv[t] = ov; ri[t] = oi; }
      }
      __syncthreads();
    }
    last = ri[0];
    __syncthreads();
  }
}

// ---------------------------------------------------------------------------
// K2: KNN (32 smallest, ascending, first-index tie-break) + diff + partials
// one block per (b,g)
// ---------------------------------------------------------------------------
__global__ __launch_bounds__(256) void k_knn(
    const float* __restrict__ pts, const float* __restrict__ centers,
    int* __restrict__ knn, float* __restrict__ diffb,
    float* __restrict__ partials) {
  __shared__ float sd[NN];
  __shared__ float rv[256];
  __shared__ float rq[256];
  __shared__ int   ri[256];
  __shared__ int   snn[SS];

  const int t = threadIdx.x;
  const int bg = blockIdx.x;
  const int b = bg >> 9;  // /GG
  const float INF = __builtin_inff();
  const float* P = pts + (size_t)b * NN * 3;

  const float cx = centers[bg * 3 + 0];
  const float cy = centers[bg * 3 + 1];
  const float cz = centers[bg * 3 + 2];

  for (int n = t; n < NN; n += 256) {
    float dx = P[n * 3 + 0] - cx;
    float dy = P[n * 3 + 1] - cy;
    float dz = P[n * 3 + 2] - cz;
    sd[n] = dx * dx + dy * dy + dz * dz;
  }
  __syncthreads();

  for (int s = 0; s < SS; ++s) {
    float bv = INF; int bi = 0x7fffffff;
    for (int n = t; n < NN; n += 256) {
      float v = sd[n];
      if (v < bv || (v == bv && n < bi)) { bv = v; bi = n; }
    }
    rv[t] = bv; ri[t] = bi;
    __syncthreads();
    for (int off = 128; off > 0; off >>= 1) {
      if (t < off) {
        float ov = rv[t + off]; int oi = ri[t + off];
        if (ov < rv[t] || (ov == rv[t] && oi < ri[t])) { rv[t] = ov; ri[t] = oi; }
      }
      __syncthreads();
    }
    if (t == 0) {
      int mi = ri[0];
      snn[s] = mi;
      sd[mi] = INF;
      knn[bg * SS + s] = mi;
    }
    __syncthreads();
  }

  // diff = neighbor - center, plus per-block sum / sumsq partials
  float v = 0.f;
  if (t < SS * 3) {
    int s = t / 3, c = t - s * 3;
    float d = P[snn[s] * 3 + c] -
              ((c == 0) ? cx : (c == 1) ? cy : cz);
    diffb[(size_t)bg * (SS * 3) + t] = d;
    v = d;
  }
  rv[t] = v; rq[t] = v * v;
  __syncthreads();
  for (int off = 128; off > 0; off >>= 1) {
    if (t < off) { rv[t] += rv[t + off]; rq[t] += rq[t + off]; }
    __syncthreads();
  }
  if (t == 0) { partials[2 * bg] = rv[0]; partials[2 * bg + 1] = rq[0]; }
}

// ---------------------------------------------------------------------------
// K3: deterministic global std reduction -> scal[0] = 1/(std+eps)
// ---------------------------------------------------------------------------
__global__ __launch_bounds__(256) void k_std(
    const float* __restrict__ partials, float* __restrict__ scal) {
  __shared__ float s1[256];
  __shared__ float s2[256];
  const int t = threadIdx.x;
  float a = 0.f, q = 0.f;
  for (int i = t; i < NGROUP; i += 256) {
    a += partials[2 * i];
    q += partials[2 * i + 1];
  }
  s1[t] = a; s2[t] = q;
  __syncthreads();
  for (int off = 128; off > 0; off >>= 1) {
    if (t < off) { s1[t] += s1[t + off]; s2[t] += s2[t + off]; }
    __syncthreads();
  }
  if (t == 0) {
    const float sz = (float)(BB * GG * SS * 3);
    float m = s1[0] / sz;
    float var = (s2[0] - sz * m * m) / (sz - 1.0f);
    float sd = sqrtf(fmaxf(var, 0.0f));
    scal[0] = 1.0f / (sd + 1e-5f);
  }
}

// ---------------------------------------------------------------------------
// K4: fused MLP per group: gather -> GEMM1(+pos embed) -> GEMM2 -> GEMM3
//     (+residual, ReLU) -> max+mean pool over S
// block = 256 threads = 8 wave32; one block per (b,g)
// ---------------------------------------------------------------------------
#define LDX 392   // 384 + 8 bf16 pad (keeps 16B align, breaks bank align)
#define LDH 200   // 192 + 8 bf16 pad

__global__ __launch_bounds__(256) void k_mlp(
    const float* __restrict__ feats,
    const float* __restrict__ b1v, const float* __restrict__ bav,
    const float* __restrict__ bbv,
    const bf16_t* __restrict__ W1b, const bf16_t* __restrict__ Wab,
    const bf16_t* __restrict__ Wbb,
    const int* __restrict__ knn, const float* __restrict__ diffb,
    const float* __restrict__ scal, float* __restrict__ outp) {
  __shared__ bf16_t sX[SS * LDX];   // X (tokens x 384) bf16; re-used for gf bf16
  __shared__ bf16_t sH[SS * LDH];   // hidden (tokens x 192) bf16
  __shared__ float  sxyz[SS * 3];   // normalized diff
  __shared__ float  sdr[64];        // 1/denom table
  __shared__ int    snn[SS];

  const int t    = threadIdx.x;
  const int lane = t & 31;
  const int w    = t >> 5;
  const int col  = lane & 15;
  const int half = lane >> 4;
  const int bg   = blockIdx.x;
  const int b    = bg >> 9;

  if (t < SS) snn[t] = knn[bg * SS + t];
  if (t < 64) sdr[t] = __powf(1000.0f, -(float)t * (1.0f / 64.0f));
  __syncthreads();

  const float inv = scal[0];
  if (t < SS * 3) sxyz[t] = diffb[(size_t)bg * (SS * 3) + t] * inv;

  // gather X = [neighbor_feats | broadcast center_feats] -> bf16 LDS
  const float* F = feats + (size_t)b * NN * DD;
  const int cidx = snn[0];
  for (int e = t; e < SS * TWO_D; e += 256) {
    int s = e / TWO_D, c = e - s * TWO_D;
    float v = (c < DD) ? F[(size_t)snn[s] * DD + c]
                       : F[(size_t)cidx * DD + (c - DD)];
    sX[s * LDX + c] = (bf16_t)v;
  }
  __syncthreads();

  // ---------------- GEMM1: (32 x 384) x (384 x 384)^T ----------------
  const int nb1 = 3 * w * 16;  // this wave owns output cols [nb1, nb1+48)
  v8f acc1[2][3];
#pragma unroll
  for (int mt = 0; mt < 2; ++mt)
#pragma unroll
    for (int i = 0; i < 3; ++i) acc1[mt][i] = vzero8();

  for (int kt = 0; kt < TWO_D / 32; ++kt) {
    const int k0 = kt * 32;
    v16bf a0 = load_frag(sX, LDX, 0, k0, lane);
    v16bf a1 = load_frag(sX, LDX, 16, k0, lane);
#pragma unroll
    for (int i = 0; i < 3; ++i) {
      v16bf bf = load_frag(W1b, TWO_D, nb1 + 16 * i, k0, lane);
      acc1[0][i] = wmma_bf16(a0, bf, acc1[0][i]);
      acc1[1][i] = wmma_bf16(a1, bf, acc1[1][i]);
    }
  }

  // epilogue: gf = relu(acc+b1); gf = (gf + pos)*pos  -- kept in registers
#pragma unroll
  for (int mt = 0; mt < 2; ++mt) {
#pragma unroll
    for (int i = 0; i < 3; ++i) {
      const int c0 = nb1 + 16 * i + col;
      const int ii = c0 >> 7;          // which coordinate (0..2)
      const int rr = c0 & 127;         // 0..127 within coord block
      const int fi = rr & 63;          // frequency index
      const bool isSin = (rr < 64);
      const float dr = sdr[fi];
      const float bc = b1v[c0];
#pragma unroll
      for (int r = 0; r < 8; ++r) {
        const int row = mt * 16 + r + 8 * half;
        float x = fmaxf(acc1[mt][i][r] + bc, 0.0f);
        float z = 100.0f * sxyz[row * 3 + ii] * dr;
        float p = isSin ? __sinf(z) : __cosf(z);
        acc1[mt][i][r] = (x + p) * p;
      }
    }
  }
  __syncthreads();  // everyone done reading sX (GEMM1)

  // write gf bf16 (re-use sX) for GEMM2 input
#pragma unroll
  for (int mt = 0; mt < 2; ++mt)
#pragma unroll
    for (int i = 0; i < 3; ++i) {
      const int c0 = nb1 + 16 * i + col;
#pragma unroll
      for (int r = 0; r < 8; ++r) {
        const int row = mt * 16 + r + 8 * half;
        sX[row * LDX + c0] = (bf16_t)acc1[mt][i][r];
      }
    }
  __syncthreads();

  // ---------------- GEMM2: (32 x 384) x (192 x 384)^T -> relu -> sH ----------------
  {
    v8f acc2[3];
#pragma unroll
    for (int i = 0; i < 3; ++i) acc2[i] = vzero8();
    const int id0 = w * 3;
    for (int kt = 0; kt < OUTC / 32; ++kt) {
      const int k0 = kt * 32;
#pragma unroll
      for (int i = 0; i < 3; ++i) {
        const int id = id0 + i;
        const int mt = id / 12, nt = id % 12;
        v16bf a  = load_frag(sX, LDX, mt * 16, k0, lane);
        v16bf bf = load_frag(Wab, OUTC, nt * 16, k0, lane);
        acc2[i] = wmma_bf16(a, bf, acc2[i]);
      }
    }
#pragma unroll
    for (int i = 0; i < 3; ++i) {
      const int id = id0 + i;
      const int mt = id / 12, nt = id % 12;
      const int c0 = nt * 16 + col;
      const float bc = bav[c0];
#pragma unroll
      for (int r = 0; r < 8; ++r) {
        const int row = mt * 16 + r + 8 * half;
        sH[row * LDH + c0] = (bf16_t)fmaxf(acc2[i][r] + bc, 0.0f);
      }
    }
  }
  __syncthreads();

  // ---------------- GEMM3 + residual + ReLU + max/mean pool ----------------
  float* outBase = outp + CENTERS_OUT + (size_t)bg * OUTC;
#pragma unroll
  for (int i = 0; i < 3; ++i) {
    const int n0 = nb1 + 16 * i;  // same columns as GEMM1 -> residual in regs
    float pmax = -__builtin_inff();
    float psum = 0.0f;
#pragma unroll
    for (int mt = 0; mt < 2; ++mt) {
      v8f acc = vzero8();
      for (int kt = 0; kt < HC / 32; ++kt) {
        const int k0 = kt * 32;
        v16bf a  = load_frag(sH, LDH, mt * 16, k0, lane);
        v16bf bf = load_frag(Wbb, HC, n0, k0, lane);
        acc = wmma_bf16(a, bf, acc);
      }
      const float bc = bbv[n0 + col];
#pragma unroll
      for (int r = 0; r < 8; ++r) {
        float v = fmaxf(acc[r] + bc + acc1[mt][i][r], 0.0f);
        pmax = fmaxf(pmax, v);
        psum += v;
      }
    }
    // combine the two lane-halves (same column, complementary rows)
    pmax = fmaxf(pmax, __shfl_xor(pmax, 16, 32));
    psum += __shfl_xor(psum, 16, 32);
    if (half == 0) outBase[n0 + col] = pmax + psum * (1.0f / 32.0f);
  }
}

// ---------------------------------------------------------------------------
// host launcher
// ---------------------------------------------------------------------------
extern "C" void kernel_launch(void* const* d_in, const int* in_sizes, int n_in,
                              void* d_out, int out_size, void* d_ws,
                              size_t ws_size, hipStream_t stream) {
  (void)in_sizes; (void)n_in; (void)out_size; (void)ws_size;
  const float* points = (const float*)d_in[0];
  const float* feats  = (const float*)d_in[1];
  const float* W1     = (const float*)d_in[2];
  const float* b1     = (const float*)d_in[3];
  const float* Wa     = (const float*)d_in[4];
  const float* ba     = (const float*)d_in[5];
  const float* Wb     = (const float*)d_in[6];
  const float* bb     = (const float*)d_in[7];
  float* out = (float*)d_out;

  char* ws = (char*)d_ws;
  size_t off = 0;
  auto alloc = [&](size_t bytes) {
    size_t r = off;
    off = (off + bytes + 255) & ~(size_t)255;
    return r;
  };
  float*  centers  = (float*)(ws + alloc((size_t)BB * GG * 3 * 4));
  int*    knn      = (int*)  (ws + alloc((size_t)NGROUP * SS * 4));
  float*  diffb    = (float*)(ws + alloc((size_t)NGROUP * SS * 3 * 4));
  float*  partials = (float*)(ws + alloc((size_t)NGROUP * 2 * 4));
  float*  scal     = (float*)(ws + alloc(256));
  bf16_t* W1b      = (bf16_t*)(ws + alloc((size_t)OUTC * TWO_D * 2));
  bf16_t* Wab      = (bf16_t*)(ws + alloc((size_t)HC * OUTC * 2));
  bf16_t* Wbb      = (bf16_t*)(ws + alloc((size_t)OUTC * HC * 2));

  k_prep<<<(OUTC * TWO_D + 255) / 256, 256, 0, stream>>>(W1, Wa, Wb, W1b, Wab,
                                                         Wbb);
  k_fps<<<BB, 256, 0, stream>>>(points, centers, out);
  k_knn<<<NGROUP, 256, 0, stream>>>(points, centers, knn, diffb, partials);
  k_std<<<1, 256, 0, stream>>>(partials, scal);
  k_mlp<<<NGROUP, 256, 0, stream>>>(feats, b1, ba, bb, W1b, Wab, Wbb, knn,
                                    diffb, scal, out);
}
